// MultiHeadedAttention_79113297592851
// MI455X (gfx1250) — matrix operand order repaired and checked by reading, not compiled, not run on hardware
//
#include <hip/hip_runtime.h>
#include <hip/hip_bf16.h>

typedef __attribute__((ext_vector_type(16))) _Float16 v16h;
typedef __attribute__((ext_vector_type(8)))  _Float16 v8h;
typedef __attribute__((ext_vector_type(4)))  _Float16 v4h;
typedef __attribute__((ext_vector_type(8)))  float    v8f;
typedef __attribute__((ext_vector_type(4)))  float    v4f;

// Problem constants
#define BB 4
#define SS 4096
#define DD 256
#define HH 8
#define DK 32
#define NTOK (BB*SS)            // 16384 tokens per tensor
#define NQ   (NTOK*DD)          // 4194304 f32 elements per tensor

// ---------------- helpers ----------------

__device__ __forceinline__ float redsum16(float v) {
    v += __shfl_xor(v, 1, 16);
    v += __shfl_xor(v, 2, 16);
    v += __shfl_xor(v, 4, 16);
    v += __shfl_xor(v, 8, 16);
    return v;
}
__device__ __forceinline__ float redmax16(float v) {
    v = fmaxf(v, __shfl_xor(v, 1, 16));
    v = fmaxf(v, __shfl_xor(v, 2, 16));
    v = fmaxf(v, __shfl_xor(v, 4, 16));
    v = fmaxf(v, __shfl_xor(v, 8, 16));
    return v;
}

// A-matrix (16x32 f16, M=16 K=32) fragment, contiguous K order (matches B):
// lanes 0-15 : row M=lane,    halfs 0-15 = K0..15
// lanes 16-31: row M=lane-16, halfs 0-15 = K16..31
// `row` points at the start of this lane's 32 contiguous K values.
__device__ __forceinline__ v16h load_a_frag(const _Float16* row, int hi) {
    v8h a0 = *(const v8h*)(row + hi * 16);
    v8h a1 = *(const v8h*)(row + hi * 16 + 8);
    v16h r;
#pragma unroll
    for (int i = 0; i < 8; ++i) { r[i] = a0[i]; r[i + 8] = a1[i]; }
    return r;
}

// gfx1250 async global->LDS copy (16B per lane), tracked with ASYNCcnt.
__device__ __forceinline__ void async_copy_b128(unsigned lds_off, const void* gaddr) {
    unsigned long long ga = (unsigned long long)(size_t)gaddr;
    asm volatile("global_load_async_to_lds_b128 %0, %1, off"
                 :: "v"(lds_off), "v"(ga) : "memory");
}
__device__ __forceinline__ void wait_async_le1() {
    asm volatile("s_wait_asynccnt 0x1" ::: "memory");
}
__device__ __forceinline__ void wait_async_le0() {
    asm volatile("s_wait_asynccnt 0x0" ::: "memory");
}

// ---------------- kernel 1: f32 -> f16 converts ----------------
// [query | key | w0] flattened, 4 elements per thread.
__global__ __launch_bounds__(256) void convert_kernel(
    const float* __restrict__ q, const float* __restrict__ k,
    const float* __restrict__ w0,
    _Float16* __restrict__ xqh, _Float16* __restrict__ xkh,
    _Float16* __restrict__ w0h) {
    int i = blockIdx.x * 256 + threadIdx.x;
    long base = (long)i * 4;
    const float* src; _Float16* dst; long off;
    if (base < (long)NQ)            { src = q;  dst = xqh; off = base; }
    else if (base < 2L * NQ)        { src = k;  dst = xkh; off = base - NQ; }
    else                            { src = w0; dst = w0h; off = base - 2L * NQ; }
    v4f a = *(const v4f*)(src + off);
    v4h r;
    r[0] = (_Float16)a[0]; r[1] = (_Float16)a[1];
    r[2] = (_Float16)a[2]; r[3] = (_Float16)a[3];
    *(v4h*)(dst + off) = r;
}

// ---------------- kernel 2: per-head scalar norms qn/kn ----------------
// one thread per (which, token, head): dot(x_row, w1_row_h) + b1[h]
__global__ __launch_bounds__(256) void norms_kernel(
    const float* __restrict__ q, const float* __restrict__ k,
    const float* __restrict__ w1, const float* __restrict__ b1,
    float* __restrict__ qnb, float* __restrict__ knb) {
    int idx = blockIdx.x * 256 + threadIdx.x;     // 0 .. 262143
    int which = idx >> 17;
    int t = (idx >> 3) & (NTOK - 1);
    int h = idx & 7;
    const v4f* xv = (const v4f*)((which ? k : q) + (long)t * DD);
    const v4f* wv = (const v4f*)(w1 + h * DD);
    float s = b1[h];
#pragma unroll 8
    for (int i = 0; i < DD / 4; ++i) {
        v4f a = xv[i], c = wv[i];
        s += a[0] * c[0] + a[1] * c[1] + a[2] * c[2] + a[3] * c[3];
    }
    (which ? knb : qnb)[t * HH + h] = s;
}

// ---------------- kernel 3: projection GEMM + normalize (WMMA) ----------------
// Block = 8 waves; wave w == head w computes C[16 tokens x 32 channels of head w]
// as two 16x16 tiles accumulated over K=256 in 8 WMMA steps each.
__global__ __launch_bounds__(256) void proj_kernel(
    const _Float16* __restrict__ xqh, const _Float16* __restrict__ xkh,
    const _Float16* __restrict__ w0h, const float* __restrict__ b0v,
    const float* __restrict__ qnb, const float* __restrict__ knb,
    _Float16* __restrict__ qh, _Float16* __restrict__ kh) {
    int lane = threadIdx.x & 31;
    int w = threadIdx.x >> 5;           // head
    int hi = lane >> 4, lo = lane & 15;
    int tile = blockIdx.x;              // 0..2047
    int which = tile >> 10;
    int token0 = (tile & 1023) * 16;

    const _Float16* xh = which ? xkh : xqh;
    const float* qn = which ? knb : qnb;
    _Float16* dst = which ? kh : qh;

    const _Float16* arow  = xh + (long)(token0 + lo) * DD;
    const _Float16* brow0 = w0h + (long)(32 * w + lo) * DD;        // channels 32w..32w+15
    const _Float16* brow1 = w0h + (long)(32 * w + 16 + lo) * DD;   // channels +16..+31

    v8f c0 = {}; v8f c1 = {};
#pragma unroll
    for (int ks = 0; ks < 8; ++ks) {
        v16h a  = load_a_frag(arow + ks * 32, hi);
        v16h b0 = *(const v16h*)(brow0 + ks * 32 + hi * 16);
        v16h b1 = *(const v16h*)(brow1 + ks * 32 + hi * 16);
        c0 = __builtin_amdgcn_wmma_f32_16x16x32_f16(false, a, false, b0, (short)0, c0, false, false);
        c1 = __builtin_amdgcn_wmma_f32_16x16x32_f16(false, a, false, b1, (short)0, c1, false, false);
    }
    float bias0 = b0v[32 * w + lo];
    float bias1 = b0v[32 * w + 16 + lo];
#pragma unroll
    for (int r = 0; r < 8; ++r) {
        float v0 = c0[r] + bias0;
        float v1 = c1[r] + bias1;
        float ss = redsum16(v0 * v0 + v1 * v1);       // ||head dir||^2 over 32 dims
        int t = token0 + r + 8 * hi;
        float sc = 10.0f * qn[t * HH + w] * rsqrtf(ss + 1e-20f);
        int bi = t >> 12, s = t & (SS - 1);
        _Float16* drow = dst + ((long)(bi * HH + w) * SS + s) * DK;
        drow[lo]      = (_Float16)(v0 * sc);
        drow[16 + lo] = (_Float16)(v1 * sc);
    }
}

// ---------------- kernel 4: flash attention (WMMA scores, async mask pipeline) ----
// Block = 8 waves = 8 heads; one 16-row q tile per block; loop over 64-key tiles.
// Mask tile is DMA'd global->LDS with a 2-deep async pipeline (ASYNCcnt).
#define MPAD 68   // ints per LDS mask row: 272B (16B aligned chunks, bank-shifted)
__global__ __launch_bounds__(256) void attn_kernel(
    const _Float16* __restrict__ qh, const _Float16* __restrict__ kh,
    const float* __restrict__ value, const int* __restrict__ mask,
    float* __restrict__ out) {
    __shared__ int   smask[2][16][MPAD];
    __shared__ float sval[64];
    __shared__ float xsh[8][16];

    int tid = threadIdx.x;
    int lane = tid & 31, w = tid >> 5, hi = lane >> 4, lo = lane & 15;
    int bi = blockIdx.x >> 8;                 // batch
    int q0 = (blockIdx.x & 255) * 16;

    const _Float16* qrow = qh + ((long)(bi * HH + w) * SS + q0 + lo) * DK;
    v16h afr = load_a_frag(qrow, hi);         // constant q fragment for whole loop
    const _Float16* kbase = kh + (long)(bi * HH + w) * SS * DK;

    float m[8], l[8], acc[8];
#pragma unroll
    for (int r = 0; r < 8; ++r) { m[r] = -1e30f; l[r] = 0.f; acc[r] = 0.f; }

    const float inv = 0.17677669529663689f;   // 1/sqrt(32)

    // each of the 256 threads owns one 16B chunk of the 16x64 mask tile
    int mrow = tid >> 4;                      // 0..15
    int mcol = (tid & 15) * 4;                // 0..60
    const int* gm = mask + ((long)(bi * SS + q0 + mrow) * SS) + mcol;
    unsigned ldst0 = (unsigned)(size_t)&smask[0][mrow][mcol];
    unsigned ldst1 = (unsigned)(size_t)&smask[1][mrow][mcol];

    async_copy_b128(ldst0, gm);               // prologue: tile 0 -> buf 0

    for (int kt = 0; kt < SS / 64; ++kt) {
        int k0 = kt * 64;
        __syncthreads();                      // prev iter done reading buf[(kt+1)&1]
        if (kt < SS / 64 - 1) {
            async_copy_b128((kt & 1) ? ldst0 : ldst1, gm + k0 + 64);
        }
        if (tid < 64) sval[tid] = value[bi * SS + k0 + tid];
        if (kt < SS / 64 - 1) wait_async_le1();   // tile kt landed; kt+1 in flight
        else                  wait_async_le0();
        __syncthreads();                      // publish LDS to all 8 waves

        const int(*mk)[MPAD] = smask[kt & 1];
        v8f s0, s1, s2, s3;
        {
            v8f z = {};
            v16h b;
            b = *(const v16h*)(kbase + (long)(k0 + lo) * DK + hi * 16);
            s0 = __builtin_amdgcn_wmma_f32_16x16x32_f16(false, afr, false, b, (short)0, z, false, false);
            b = *(const v16h*)(kbase + (long)(k0 + 16 + lo) * DK + hi * 16);
            s1 = __builtin_amdgcn_wmma_f32_16x16x32_f16(false, afr, false, b, (short)0, z, false, false);
            b = *(const v16h*)(kbase + (long)(k0 + 32 + lo) * DK + hi * 16);
            s2 = __builtin_amdgcn_wmma_f32_16x16x32_f16(false, afr, false, b, (short)0, z, false, false);
            b = *(const v16h*)(kbase + (long)(k0 + 48 + lo) * DK + hi * 16);
            s3 = __builtin_amdgcn_wmma_f32_16x16x32_f16(false, afr, false, b, (short)0, z, false, false);
        }
        float v0 = sval[lo], v1 = sval[16 + lo], v2 = sval[32 + lo], v3 = sval[48 + lo];
#pragma unroll
        for (int r = 0; r < 8; ++r) {
            int M = r + 8 * hi;
            float t0 = mk[M][lo]      ? s0[r] * inv : -1e9f;
            float t1 = mk[M][16 + lo] ? s1[r] * inv : -1e9f;
            float t2 = mk[M][32 + lo] ? s2[r] * inv : -1e9f;
            float t3 = mk[M][48 + lo] ? s3[r] * inv : -1e9f;
            float rm = redmax16(fmaxf(fmaxf(t0, t1), fmaxf(t2, t3)));
            float newm = fmaxf(m[r], rm);
            float corr = __expf(m[r] - newm);
            float p0 = __expf(t0 - newm), p1 = __expf(t1 - newm);
            float p2 = __expf(t2 - newm), p3 = __expf(t3 - newm);
            float ps = redsum16(p0 + p1 + p2 + p3);
            float pv = redsum16(p0 * v0 + p1 * v1 + p2 * v2 + p3 * v3);
            l[r] = l[r] * corr + ps;
            acc[r] = acc[r] * corr + pv;
            m[r] = newm;
        }
    }
    if (lo == 0) {
#pragma unroll
        for (int r = 0; r < 8; ++r) xsh[w][r + 8 * hi] = acc[r] / l[r];
    }
    __syncthreads();
    if (tid < 16) {
        float s = 0.f;
#pragma unroll
        for (int h = 0; h < 8; ++h) s += xsh[h][tid];
        out[bi * SS + q0 + tid] = s * 0.125f;   // mean over heads
    }
}

// ---------------- launch ----------------
extern "C" void kernel_launch(void* const* d_in, const int* in_sizes, int n_in,
                              void* d_out, int out_size, void* d_ws, size_t ws_size,
                              hipStream_t stream) {
    const float* query = (const float*)d_in[0];
    const float* key   = (const float*)d_in[1];
    const float* value = (const float*)d_in[2];
    const int*   mask  = (const int*)d_in[3];
    const float* w0    = (const float*)d_in[4];
    const float* b0    = (const float*)d_in[5];
    const float* w1    = (const float*)d_in[6];
    const float* b1    = (const float*)d_in[7];
    float* out = (float*)d_out;

    char* ws = (char*)d_ws;
    _Float16* xqh = (_Float16*)(ws);                       //  8 MB  [token][256] f16
    _Float16* xkh = (_Float16*)(ws + 8388608);             //  8 MB
    _Float16* w0h = (_Float16*)(ws + 16777216);            // 128 KB [256][256] f16
    _Float16* qhb = (_Float16*)(ws + 16908288);            //  8 MB  [B][H][S][32] f16
    _Float16* khb = (_Float16*)(ws + 25296896);            //  8 MB
    float*    qnb = (float*)(ws + 33685504);               // 512 KB [token][H] f32
    float*    knb = (float*)(ws + 34209792);               // 512 KB

    convert_kernel<<<8256, 256, 0, stream>>>(query, key, w0, xqh, xkh, w0h);
    norms_kernel<<<1024, 256, 0, stream>>>(query, key, w1, b1, qnb, knb);
    proj_kernel<<<2048, 256, 0, stream>>>(xqh, xkh, w0h, b0, qnb, knb, qhb, khb);
    attn_kernel<<<BB * (SS / 16), 256, 0, stream>>>(qhb, khb, value, mask, out);
}
